// DKVMN_75574244540555
// MI455X (gfx1250) — compile-verified, matrix operand order checked by reading, not run on hardware
//
#include <hip/hip_runtime.h>
#include <hip/hip_bf16.h>
#include <math.h>

// ---------------------------------------------------------------------------
// CDNA5 (gfx1250) DKVMN forward. wave32, WMMA bf16 16x16x32 for all GEMMs.
// ---------------------------------------------------------------------------

typedef __attribute__((ext_vector_type(16))) __bf16        v16bf;
typedef __attribute__((ext_vector_type(8)))  float         v8f;
typedef __attribute__((ext_vector_type(4)))  unsigned int  v4u;
typedef __attribute__((ext_vector_type(4)))  float         v4f;

union FragBF {
    v16bf v;
    v4u   q[2];
};

__device__ __forceinline__ unsigned short f2bf(float f) {
    unsigned int u = __float_as_uint(f);
    u += 0x7FFFu + ((u >> 16) & 1u);   // round-to-nearest-even
    return (unsigned short)(u >> 16);
}

__device__ __forceinline__ v8f wmma_bf16(const FragBF& a, const FragBF& b, v8f c) {
    // (neg_a, A, neg_b, B, c_mod, C, reuse_a, reuse_b)
    return __builtin_amdgcn_wmma_f32_16x16x32_bf16(false, a.v, false, b.v,
                                                   (short)0, c, false, false);
}

// ---------------------------------------------------------------------------
// Kernel 1: convert + transpose weights to bf16 [N][K] layout.
// Regions: W1t(128x512), W2t(128x512), Wet(128x128), Wat(128x128),
//          Wft(128x256), Mkt(64x128, rows 50..63 zero). Total 204800 elems.
// ---------------------------------------------------------------------------
__global__ void __launch_bounds__(256)
prep_weights(const float* __restrict__ W1, const float* __restrict__ W2,
             const float* __restrict__ We, const float* __restrict__ Wa,
             const float* __restrict__ Wf, const float* __restrict__ Mk,
             unsigned short* __restrict__ W1t, unsigned short* __restrict__ W2t,
             unsigned short* __restrict__ Wet, unsigned short* __restrict__ Wat,
             unsigned short* __restrict__ Wft, unsigned short* __restrict__ Mkt)
{
    int i = blockIdx.x * 256 + threadIdx.x;
    if (i < 65536) { int n = i >> 9, k = i & 511; W1t[i] = f2bf(W1[k * 128 + n]); return; }
    i -= 65536;
    if (i < 65536) { int n = i >> 9, k = i & 511; W2t[i] = f2bf(W2[k * 128 + n]); return; }
    i -= 65536;
    if (i < 16384) { int n = i >> 7, k = i & 127; Wet[i] = f2bf(We[k * 128 + n]); return; }
    i -= 16384;
    if (i < 16384) { int n = i >> 7, k = i & 127; Wat[i] = f2bf(Wa[k * 128 + n]); return; }
    i -= 16384;
    if (i < 32768) { int n = i >> 8, k = i & 255; Wft[i] = f2bf(Wf[k * 128 + n]); return; }
    i -= 32768;
    if (i < 8192)  { int m = i >> 7;              Mkt[i] = (m < 50) ? f2bf(Mk[i]) : (unsigned short)0; }
}

// ---------------------------------------------------------------------------
// Kernel 2: gathered embedding GEMM: out[16rows x 128] = emb[id] @ Wt^T + bias
// K = 512. 256 threads = 8 waves, one 16x16 N-tile per wave.
// ---------------------------------------------------------------------------
__global__ void __launch_bounds__(256)
gather_gemm_kv(const int* __restrict__ q, const int* __restrict__ rr, int use_r,
               const float* __restrict__ emb, const unsigned short* __restrict__ Bt,
               const float* __restrict__ bias, float* __restrict__ out)
{
    __shared__ __align__(16) unsigned short As[16 * 520];   // 16 rows x 512 bf16, +8 pad
    const int tid  = threadIdx.x;
    const int row0 = blockIdx.x * 16;

    {   // stage A: gather + f32->bf16
        const int rl  = tid >> 4;
        const int c16 = tid & 15;
        const int row = row0 + rl;
        int id = q[row];
        if (use_r) id += 10000 * rr[row];
        const v4f* src = (const v4f*)(emb + (long long)id * 512);
        unsigned short* dstRow = &As[rl * 520];
#pragma unroll
        for (int j = 0; j < 8; ++j) {
            const int c4 = c16 + j * 16;
            v4f f = src[c4];
            unsigned short* d = dstRow + c4 * 4;
            d[0] = f2bf(f.x); d[1] = f2bf(f.y); d[2] = f2bf(f.z); d[3] = f2bf(f.w);
        }
    }
    __syncthreads();

    const int lane = tid & 31;
    const int wave = tid >> 5;
    const int n    = lane & 15;          // col for B, row for A
    const int half = lane >> 4;
    const unsigned short* brow = Bt + (size_t)(wave * 16 + n) * 512;

    v8f acc = {0.f, 0.f, 0.f, 0.f, 0.f, 0.f, 0.f, 0.f};
#pragma unroll 4
    for (int kk = 0; kk < 512; kk += 32) {
        const int kb = kk + half * 8;
        FragBF a, b;
        a.q[0] = *(const v4u*)&As[n * 520 + kb];
        a.q[1] = *(const v4u*)&As[n * 520 + kb + 16];
        b.q[0] = *(const v4u*)&brow[kb];
        b.q[1] = *(const v4u*)&brow[kb + 16];
        __builtin_prefetch(&brow[kb + 64], 0, 3);
        acc = wmma_bf16(a, b, acc);
    }

    const int col = wave * 16 + n;
    const float bv = bias[col];
#pragma unroll
    for (int r8 = 0; r8 < 8; ++r8) {
        const int m = r8 + half * 8;
        out[(long long)(row0 + m) * 128 + col] = acc[r8] + bv;
    }
}

// ---------------------------------------------------------------------------
// Kernel 3: out = act(A @ Bt^T + bias), A [BL,128] f32, Bt [128][128] bf16.
// act: 1 = sigmoid, 2 = tanh.
// ---------------------------------------------------------------------------
__global__ void __launch_bounds__(256)
gemm128_act(const float* __restrict__ A, const unsigned short* __restrict__ Bt,
            const float* __restrict__ bias, float* __restrict__ out, int act)
{
    __shared__ __align__(16) unsigned short As[16 * 136];
    const int tid  = threadIdx.x;
    const int row0 = blockIdx.x * 16;
    {
        const int rl  = tid >> 4;
        const int c16 = tid & 15;
        const v4f* src = (const v4f*)(A + (long long)(row0 + rl) * 128);
        unsigned short* dstRow = &As[rl * 136];
#pragma unroll
        for (int j = 0; j < 2; ++j) {
            const int c4 = c16 + j * 16;
            v4f f = src[c4];
            unsigned short* d = dstRow + c4 * 4;
            d[0] = f2bf(f.x); d[1] = f2bf(f.y); d[2] = f2bf(f.z); d[3] = f2bf(f.w);
        }
    }
    __syncthreads();

    const int lane = tid & 31;
    const int wave = tid >> 5;
    const int n    = lane & 15;
    const int half = lane >> 4;
    const unsigned short* brow = Bt + (size_t)(wave * 16 + n) * 128;

    v8f acc = {0.f, 0.f, 0.f, 0.f, 0.f, 0.f, 0.f, 0.f};
#pragma unroll
    for (int kk = 0; kk < 128; kk += 32) {
        const int kb = kk + half * 8;
        FragBF a, b;
        a.q[0] = *(const v4u*)&As[n * 136 + kb];
        a.q[1] = *(const v4u*)&As[n * 136 + kb + 16];
        b.q[0] = *(const v4u*)&brow[kb];
        b.q[1] = *(const v4u*)&brow[kb + 16];
        acc = wmma_bf16(a, b, acc);
    }

    const int col = wave * 16 + n;
    const float bv = bias[col];
#pragma unroll
    for (int r8 = 0; r8 < 8; ++r8) {
        float v = acc[r8] + bv;
        if (act == 1)      v = 1.f / (1.f + __expf(-v));
        else if (act == 2) v = tanhf(v);
        out[(long long)(row0 + r8 + half * 8) * 128 + col] = v;
    }
}

// ---------------------------------------------------------------------------
// Kernel 4: w = softmax(k @ Mk^T) over the first 50 of 64 cols; cols 50..63 = 0.
// 128 threads = 4 waves (N = 64 padded), K = 128.
// ---------------------------------------------------------------------------
__global__ void __launch_bounds__(128)
gemm_w_softmax(const float* __restrict__ Kmat, const unsigned short* __restrict__ Mkt,
               float* __restrict__ Wout)
{
    __shared__ __align__(16) unsigned short As[16 * 136];
    __shared__ float Lg[16 * 68];
    const int tid  = threadIdx.x;
    const int row0 = blockIdx.x * 16;
    {
        const int rl = tid >> 3;
        const int c8 = tid & 7;
        const v4f* src = (const v4f*)(Kmat + (long long)(row0 + rl) * 128);
        unsigned short* dstRow = &As[rl * 136];
#pragma unroll
        for (int j = 0; j < 4; ++j) {
            const int c4 = c8 + j * 8;
            v4f f = src[c4];
            unsigned short* d = dstRow + c4 * 4;
            d[0] = f2bf(f.x); d[1] = f2bf(f.y); d[2] = f2bf(f.z); d[3] = f2bf(f.w);
        }
    }
    __syncthreads();

    const int lane = tid & 31;
    const int wave = tid >> 5;
    const int n    = lane & 15;
    const int half = lane >> 4;
    const unsigned short* brow = Mkt + (size_t)(wave * 16 + n) * 128;

    v8f acc = {0.f, 0.f, 0.f, 0.f, 0.f, 0.f, 0.f, 0.f};
#pragma unroll
    for (int kk = 0; kk < 128; kk += 32) {
        const int kb = kk + half * 8;
        FragBF a, b;
        a.q[0] = *(const v4u*)&As[n * 136 + kb];
        a.q[1] = *(const v4u*)&As[n * 136 + kb + 16];
        b.q[0] = *(const v4u*)&brow[kb];
        b.q[1] = *(const v4u*)&brow[kb + 16];
        acc = wmma_bf16(a, b, acc);
    }
#pragma unroll
    for (int r8 = 0; r8 < 8; ++r8)
        Lg[(r8 + half * 8) * 68 + wave * 16 + n] = acc[r8];
    __syncthreads();

    if (tid < 16) {
        const float* lrow = &Lg[tid * 68];
        float mx = lrow[0];
        for (int m = 1; m < 50; ++m) mx = fmaxf(mx, lrow[m]);
        float s = 0.f;
        for (int m = 0; m < 50; ++m) s += __expf(lrow[m] - mx);
        const float inv = 1.f / s;
        float* wrow = &Wout[(long long)(row0 + tid) * 64];
        for (int m = 0; m < 50; ++m) wrow[m] = __expf(lrow[m] - mx) * inv;
        for (int m = 50; m < 64; ++m) wrow[m] = 0.f;
    }
}

// ---------------------------------------------------------------------------
// Kernel 5: sequential memory scan. One block per batch, one lane per D-column,
// 50-entry memory column kept in registers. L = 200 steps.
// ---------------------------------------------------------------------------
__global__ void __launch_bounds__(128)
dkvmn_scan(const float* __restrict__ Wm, const float* __restrict__ E,
           const float* __restrict__ A, const float* __restrict__ Mv0,
           float* __restrict__ Reads)
{
    __shared__ float ws[64];
    const int b = blockIdx.x;
    const int d = threadIdx.x;

    float Mv[50];
#pragma unroll
    for (int m = 0; m < 50; ++m) Mv[m] = Mv0[m * 128 + d];

    for (int t = 0; t < 200; ++t) {
        const long long idx = (long long)b * 200 + t;
        if (d < 64) ws[d] = Wm[idx * 64 + d];
        const float ed = E[idx * 128 + d];
        const float ad = A[idx * 128 + d];
        __syncthreads();
        float rt = 0.f;
#pragma unroll
        for (int m = 0; m < 50; ++m) {
            const float wm = ws[m];
            rt += wm * Mv[m];                                   // read BEFORE update
            Mv[m] = Mv[m] * (1.f - wm * ed) + wm * ad;
        }
        Reads[idx * 128 + d] = rt;
        __syncthreads();
    }
}

// ---------------------------------------------------------------------------
// Kernel 6: f = tanh([reads,k] @ Wf^T + bf), p = sigmoid(f @ Wp + bp), fused.
// K = 256, 8 waves.
// ---------------------------------------------------------------------------
__global__ void __launch_bounds__(256)
gemm_f_p(const float* __restrict__ Reads, const float* __restrict__ Kmat,
         const unsigned short* __restrict__ Wft, const float* __restrict__ bf_,
         const float* __restrict__ Wp, const float* __restrict__ bp,
         float* __restrict__ P)
{
    __shared__ __align__(16) unsigned short As[16 * 264];
    __shared__ float Fs[16 * 132];
    const int tid  = threadIdx.x;
    const int row0 = blockIdx.x * 16;
    {
        const int rl  = tid >> 4;
        const int c16 = tid & 15;
        const v4f* s1 = (const v4f*)(Reads + (long long)(row0 + rl) * 128);
        const v4f* s2 = (const v4f*)(Kmat  + (long long)(row0 + rl) * 128);
        unsigned short* dstRow = &As[rl * 264];
#pragma unroll
        for (int j = 0; j < 2; ++j) {
            const int c4 = c16 + j * 16;
            v4f f = s1[c4];
            unsigned short* d = dstRow + c4 * 4;
            d[0] = f2bf(f.x); d[1] = f2bf(f.y); d[2] = f2bf(f.z); d[3] = f2bf(f.w);
            v4f g = s2[c4];
            unsigned short* d2 = dstRow + 128 + c4 * 4;
            d2[0] = f2bf(g.x); d2[1] = f2bf(g.y); d2[2] = f2bf(g.z); d2[3] = f2bf(g.w);
        }
    }
    __syncthreads();

    const int lane = tid & 31;
    const int wave = tid >> 5;
    const int n    = lane & 15;
    const int half = lane >> 4;
    const unsigned short* brow = Wft + (size_t)(wave * 16 + n) * 256;

    v8f acc = {0.f, 0.f, 0.f, 0.f, 0.f, 0.f, 0.f, 0.f};
#pragma unroll
    for (int kk = 0; kk < 256; kk += 32) {
        const int kb = kk + half * 8;
        FragBF a, b;
        a.q[0] = *(const v4u*)&As[n * 264 + kb];
        a.q[1] = *(const v4u*)&As[n * 264 + kb + 16];
        b.q[0] = *(const v4u*)&brow[kb];
        b.q[1] = *(const v4u*)&brow[kb + 16];
        acc = wmma_bf16(a, b, acc);
    }

    const int col = wave * 16 + n;
    const float bv = bf_[col];
#pragma unroll
    for (int r8 = 0; r8 < 8; ++r8)
        Fs[(r8 + half * 8) * 132 + col] = tanhf(acc[r8] + bv);
    __syncthreads();

    if (tid < 16) {
        const float* frow = &Fs[tid * 132];
        float s = bp[0];
        for (int dd = 0; dd < 128; ++dd) s += frow[dd] * Wp[dd];
        P[row0 + tid] = 1.f / (1.f + __expf(-s));
    }
}

// ---------------------------------------------------------------------------
extern "C" void kernel_launch(void* const* d_in, const int* in_sizes, int n_in,
                              void* d_out, int out_size, void* d_ws, size_t ws_size,
                              hipStream_t stream)
{
    const int*   q     = (const int*)  d_in[0];
    const int*   r     = (const int*)  d_in[1];
    /* d_in[2] = diff (unused by reference path) */
    const float* k_emb = (const float*)d_in[3];
    const float* v_emb = (const float*)d_in[4];
    const float* W1    = (const float*)d_in[5];
    const float* b1    = (const float*)d_in[6];
    const float* W2    = (const float*)d_in[7];
    const float* b2    = (const float*)d_in[8];
    const float* Mk    = (const float*)d_in[9];
    const float* Mv0   = (const float*)d_in[10];
    const float* We    = (const float*)d_in[11];
    const float* be    = (const float*)d_in[12];
    const float* Wa    = (const float*)d_in[13];
    const float* ba    = (const float*)d_in[14];
    const float* Wf    = (const float*)d_in[15];
    const float* bf_   = (const float*)d_in[16];
    const float* Wp    = (const float*)d_in[17];
    const float* bp    = (const float*)d_in[18];
    float* out = (float*)d_out;

    char* ws = (char*)d_ws;
    unsigned short* W1t = (unsigned short*)(ws + 0);
    unsigned short* W2t = (unsigned short*)(ws + 131072);
    unsigned short* Wet = (unsigned short*)(ws + 262144);
    unsigned short* Wat = (unsigned short*)(ws + 294912);
    unsigned short* Wft = (unsigned short*)(ws + 327680);
    unsigned short* Mkt = (unsigned short*)(ws + 393216);
    float* Kb = (float*)(ws + 409600);            // [12800,128]
    float* Vb = (float*)(ws + 409600 + 6553600);  // [12800,128]
    float* Eb = (float*)(ws + 409600 + 2 * 6553600);
    float* Ab = (float*)(ws + 409600 + 3 * 6553600);
    float* Wb = (float*)(ws + 409600 + 4 * 6553600);              // [12800,64]
    float* Rb = (float*)(ws + 409600 + 4 * 6553600 + 3276800);    // [12800,128]

    prep_weights  <<<800, 256, 0, stream>>>(W1, W2, We, Wa, Wf, Mk,
                                            W1t, W2t, Wet, Wat, Wft, Mkt);
    gather_gemm_kv<<<800, 256, 0, stream>>>(q, r, 0, k_emb, W1t, b1, Kb);
    gather_gemm_kv<<<800, 256, 0, stream>>>(q, r, 1, v_emb, W2t, b2, Vb);
    gemm_w_softmax<<<800, 128, 0, stream>>>(Kb, Mkt, Wb);
    gemm128_act   <<<800, 256, 0, stream>>>(Vb, Wet, be, Eb, 1);  // sigmoid
    gemm128_act   <<<800, 256, 0, stream>>>(Vb, Wat, ba, Ab, 2);  // tanh
    dkvmn_scan    <<<64, 128, 0, stream>>>(Wb, Eb, Ab, Mv0, Rb);
    gemm_f_p      <<<800, 256, 0, stream>>>(Rb, Kb, Wft, bf_, Wp, bp, out);
}
